// RNNModel_53549652246969
// MI455X (gfx1250) — matrix operand order repaired and checked by reading, not compiled, not run on hardware
//
#include <hip/hip_runtime.h>
#include <hip/hip_bf16.h>

// ---------------------------------------------------------------------------
// RNN (3-layer ReLU) for MI455X / gfx1250.
//   wave32 + v_wmma_f32_16x16x32_bf16, TDM weight staging (256 KB LDS/WG),
//   persistent scan kernel with cluster-barrier fast path.
//   B=64, T=512, D_IN=256, H=1024, L=3, D_OUT=128
// ---------------------------------------------------------------------------

typedef __attribute__((ext_vector_type(16))) __bf16       v16bf;
typedef __attribute__((ext_vector_type(8)))  float        v8f;
typedef __attribute__((ext_vector_type(4)))  unsigned int u32x4;
typedef __attribute__((ext_vector_type(8)))  int          i32x8;
typedef __attribute__((ext_vector_type(4)))  int          i32x4;
typedef __attribute__((ext_vector_type(4)))  float        f32x4;

#define BB   64
#define TT   512
#define DIN  256
#define HH   1024
#define DOUT 128

#define RNN_BLOCKS  8          // scan workgroups (fits one 16-WG cluster)
#define RNN_THREADS 512        // 16 waves; WG owns 128 columns = 256 KB LDS
#define RNN_COLS    128        // columns of W_hh per workgroup

// hwreg(IB_STS2, offset=6, size=4) -> Cluster_ID (0 = not in a cluster)
#define HWREG_IB_STS2_CLUSTER_ID (28 | (6 << 6) | ((4 - 1) << 11))

union FragBF {
    v16bf v;
    u32x4 q[2];
};

__device__ inline v8f vzero8() {
    v8f z = {0.f, 0.f, 0.f, 0.f, 0.f, 0.f, 0.f, 0.f};
    return z;
}

// A-matrix fragment (16x32 bf16, MxK). ISA layout:
//   lanes 0-15:  m=lane,    v0-3 = K 0..7,  v4-7 = K 16..23
//   lanes 16-31: m=lane-16, v0-3 = K 8..15, v4-7 = K 24..31
__device__ inline v16bf load_a_frag(const __bf16* __restrict__ rowPtr, int kc, int lane) {
    int kbase = kc * 32 + ((lane & 16) ? 8 : 0);
    FragBF f;
    f.q[0] = *(const u32x4*)(rowPtr + kbase);
    f.q[1] = *(const u32x4*)(rowPtr + kbase + 16);
    return f.v;
}

// B-matrix fragment (32x16 bf16, KxN). ISA layout:
//   lanes 0-15:  n=lane,    v0-3 = K 0..7,  v4-7 = K 16..23
//   lanes 16-31: n=lane-16, v0-3 = K 8..15, v4-7 = K 24..31
__device__ inline v16bf load_b_frag(const __bf16* __restrict__ rowPtr, int kc, int lane) {
    int kbase = kc * 32 + ((lane & 16) ? 8 : 0);
    FragBF f;
    f.q[0] = *(const u32x4*)(rowPtr + kbase);
    f.q[1] = *(const u32x4*)(rowPtr + kbase + 16);
    return f.v;
}

__device__ inline v8f wmma_bf16(v16bf a, v16bf b, v8f c) {
    return __builtin_amdgcn_wmma_f32_16x16x32_bf16(false, a, false, b, (short)0, c,
                                                   false, false);
}

// ---------------------------------------------------------------------------
// fp32 -> bf16 conversion, 4 elements / thread
// ---------------------------------------------------------------------------
__global__ void cvt_f32_bf16_x4(const float* __restrict__ in,
                                __bf16* __restrict__ out, int n4) {
    int i = blockIdx.x * blockDim.x + threadIdx.x;
    if (i < n4) {
        f32x4 v = ((const f32x4*)in)[i];
        __bf16* o = out + (size_t)i * 4;
        o[0] = (__bf16)v.x;
        o[1] = (__bf16)v.y;
        o[2] = (__bf16)v.z;
        o[3] = (__bf16)v.w;
    }
}

// ---------------------------------------------------------------------------
// Input-projection GEMM:  C[M,N] = A[M,K](bf16) * W[N,K]^T(bf16) + bi[n]+bh[n]
// One wave computes a 16(M) x 64(N) strip, register double-buffered so chunk
// kc+1 loads are in flight while chunk kc's WMMAs execute.
// ---------------------------------------------------------------------------
__global__ void __launch_bounds__(256)
gemm_bf16_bias(const __bf16* __restrict__ A,
               const __bf16* __restrict__ W,
               const float* __restrict__ bi,
               const float* __restrict__ bh,
               float* __restrict__ C,
               int M, int N, int K)
{
    const int lane = threadIdx.x & 31;
    const int wave = threadIdx.x >> 5;
    const int strip = blockIdx.x * 8 + wave;
    const int nStrips = N >> 6;            // strips of 64 columns
    const int mTile = strip / nStrips;
    const int nOff  = (strip % nStrips) << 6;
    if (mTile * 16 >= M) return;

    const int m = mTile * 16 + (lane & 15);
    const __bf16* aRow = A + (size_t)m * K;
    const __bf16* wRow[4];
#pragma unroll
    for (int j = 0; j < 4; ++j)
        wRow[j] = W + (size_t)(nOff + j * 16 + (lane & 15)) * K;

    v8f acc[4];
    acc[0] = vzero8(); acc[1] = vzero8(); acc[2] = vzero8(); acc[3] = vzero8();

    const int nChunks = K >> 5;

    // Preload chunk 0
    v16bf af = load_a_frag(aRow, 0, lane);
    v16bf bf0 = load_b_frag(wRow[0], 0, lane);
    v16bf bf1 = load_b_frag(wRow[1], 0, lane);
    v16bf bf2 = load_b_frag(wRow[2], 0, lane);
    v16bf bf3 = load_b_frag(wRow[3], 0, lane);

    for (int kc = 0; kc < nChunks; ++kc) {
        const int kn = (kc + 1 < nChunks) ? kc + 1 : kc;  // last iter: dummy reload
        __builtin_prefetch(aRow + (kn + 1) * 32, 0, 3);
        v16bf afN  = load_a_frag(aRow, kn, lane);
        v16bf bf0N = load_b_frag(wRow[0], kn, lane);
        v16bf bf1N = load_b_frag(wRow[1], kn, lane);
        v16bf bf2N = load_b_frag(wRow[2], kn, lane);
        v16bf bf3N = load_b_frag(wRow[3], kn, lane);

        acc[0] = wmma_bf16(af, bf0, acc[0]);
        acc[1] = wmma_bf16(af, bf1, acc[1]);
        acc[2] = wmma_bf16(af, bf2, acc[2]);
        acc[3] = wmma_bf16(af, bf3, acc[3]);

        af = afN; bf0 = bf0N; bf1 = bf1N; bf2 = bf2N; bf3 = bf3N;
    }

    // C/D layout: VGPR r -> row (r + (lane<16?0:8)), col = lane%16
    const int rowBase = mTile * 16 + ((lane & 16) ? 8 : 0);
#pragma unroll
    for (int j = 0; j < 4; ++j) {
        int n = nOff + j * 16 + (lane & 15);
        float bias = bi[n] + bh[n];
#pragma unroll
        for (int r = 0; r < 8; ++r) {
            C[(size_t)(rowBase + r) * N + n] = acc[j][r] + bias;
        }
    }
}

// ---------------------------------------------------------------------------
// Grid-wide sync for the persistent scan kernel.
// Fast path: if dispatched as a workgroup cluster (IB_STS2.CLUSTER_ID != 0,
// zero-initialized on normal dispatch), wave 0 uses the hardware cluster
// split-barrier (s_barrier_signal -3 / s_barrier_wait -3). Fallback: one
// device-scope atomic arrival per WG + spin.
// ---------------------------------------------------------------------------
__device__ inline void grid_sync(unsigned int* counter, unsigned int target) {
    __threadfence();
    __syncthreads();
    if (__builtin_amdgcn_s_getreg(HWREG_IB_STS2_CLUSTER_ID) != 0) {
        if (threadIdx.x < 32) {             // one wave per WG signals + waits
            __builtin_amdgcn_s_cluster_barrier();
        }
    } else if (threadIdx.x == 0) {
        __hip_atomic_fetch_add(counter, 1u, __ATOMIC_ACQ_REL,
                               __HIP_MEMORY_SCOPE_AGENT);
        while (__hip_atomic_load(counter, __ATOMIC_ACQUIRE,
                                 __HIP_MEMORY_SCOPE_AGENT) < target) {
            __builtin_amdgcn_s_sleep(2);
        }
    }
    __syncthreads();
}

// ---------------------------------------------------------------------------
// Persistent recurrence kernel (one launch per layer).
//   h_t = relu(pre[:,t,:] + h_{t-1} @ W_hh^T),  h_0 = 0
// 8 WGs x 512 threads; WG g owns columns [g*128, g*128+128) and keeps those
// 128 rows of W_hh resident in LDS (256 KB — CDNA5's 320 KB/WGP makes this
// possible) for all 512 steps, staged via the Tensor Data Mover.
// ---------------------------------------------------------------------------
__global__ void __launch_bounds__(RNN_THREADS)
rnn_scan(const float* __restrict__ pre,     // [B,T,H] f32
         const __bf16* __restrict__ Whh,    // [H,H]  bf16
         __bf16* __restrict__ seq,          // [B,T,H] bf16 (output)
         unsigned int* __restrict__ counter)
{
    extern __shared__ __bf16 sW[];          // RNN_COLS * 1024 bf16 = 256 KB

    const int g    = blockIdx.x;
    const int lane = threadIdx.x & 31;
    const int wave = threadIdx.x >> 5;

#if __has_builtin(__builtin_amdgcn_tensor_load_to_lds)
    // TDM: DMA the 128x1024 bf16 tile (rows g*128.. of W_hh) into LDS.
    if (threadIdx.x < 32) {                 // wave 0 issues; EXEC ignored by TDM
        unsigned long long ga =
            (unsigned long long)(const void*)(Whh + (size_t)g * RNN_COLS * HH);
        u32x4 g0;
        g0.x = 0x1u;                                        // count=1, no gather
        g0.y = 0u;                                          // lds_addr = 0
        g0.z = (unsigned int)ga;                            // global_addr[31:0]
        g0.w = (unsigned int)((ga >> 32) & 0x01FFFFFFu)     // global_addr[56:32]
               | (2u << 30);                                // type = 2 (image)
        i32x8 g1;
        g1[0] = 0x10000;              // workgroup_mask=0, data_size=1 (2 bytes)
        g1[1] = (int)(1024u << 16);   // tensor_dim0[15:0]=1024 in bits[63:48]
        g1[2] = (int)((unsigned)RNN_COLS << 16); // tensor_dim1[15:0]=128
        g1[3] = (int)(1024u << 16);   // tile_dim0 = 1024
        g1[4] = RNN_COLS;             // tile_dim1 = 128, tile_dim2 = 0
        g1[5] = 1024;                 // tensor_dim0_stride[31:0] = 1024
        g1[6] = 0;
        g1[7] = 0;
        i32x4 gz = {0, 0, 0, 0};
#if defined(__clang_major__) && __clang_major__ >= 23
        i32x8 gz8 = {0, 0, 0, 0, 0, 0, 0, 0};
        __builtin_amdgcn_tensor_load_to_lds(g0, g1, gz, gz, gz8, 0);
#else
        __builtin_amdgcn_tensor_load_to_lds(g0, g1, gz, gz, 0);
#endif
        __builtin_amdgcn_s_wait_tensorcnt((short)0);
    }
    __syncthreads();
#else
    {
        const u32x4* src = (const u32x4*)(Whh + (size_t)g * RNN_COLS * HH);
        u32x4* dst = (u32x4*)sW;
        for (int i = threadIdx.x; i < (RNN_COLS * HH) / 8; i += RNN_THREADS)
            dst[i] = src[i];
    }
    __syncthreads();
#endif

    const int bb      = wave & 3;           // b-block (16 rows each), 4 blocks
    const int nlBase  = (wave >> 2) * 32;   // local column base within 128
    const int nBase   = g * RNN_COLS + nlBase;
    const int bRow    = bb * 16 + (lane & 15);
    const int rowBase = bb * 16 + ((lane & 16) ? 8 : 0);
    const __bf16* wRow0 = sW + (size_t)(nlBase + (lane & 15)) * HH;
    const __bf16* wRow1 = sW + (size_t)(nlBase + 16 + (lane & 15)) * HH;

    for (int t = 0; t < TT; ++t) {
        v8f acc[2];
        acc[0] = vzero8(); acc[1] = vzero8();

        if (t > 0) {
            const __bf16* aRow = seq + ((size_t)bRow * TT + (t - 1)) * HH;
            // double-buffered K loop (A from global, B from LDS)
            v16bf af  = load_a_frag(aRow, 0, lane);
            v16bf bf0 = load_b_frag(wRow0, 0, lane);
            v16bf bf1 = load_b_frag(wRow1, 0, lane);
            for (int kc = 0; kc < HH / 32; ++kc) {
                const int kn = (kc + 1 < HH / 32) ? kc + 1 : kc;
                v16bf afN  = load_a_frag(aRow, kn, lane);
                v16bf bf0N = load_b_frag(wRow0, kn, lane);
                v16bf bf1N = load_b_frag(wRow1, kn, lane);
                acc[0] = wmma_bf16(af, bf0, acc[0]);
                acc[1] = wmma_bf16(af, bf1, acc[1]);
                af = afN; bf0 = bf0N; bf1 = bf1N;
            }
        }

        // Fused epilogue: + pre, ReLU, f32 -> bf16 store
#pragma unroll
        for (int j = 0; j < 2; ++j) {
            int n = nBase + j * 16 + (lane & 15);
#pragma unroll
            for (int r = 0; r < 8; ++r) {
                size_t idx = ((size_t)(rowBase + r) * TT + t) * HH + n;
                float v = acc[j][r] + pre[idx];
                v = v > 0.f ? v : 0.f;
                seq[idx] = (__bf16)v;
            }
        }

        grid_sync(counter, (unsigned int)(t + 1) * RNN_BLOCKS);
    }
}

// ---------------------------------------------------------------------------
// Final FC on last timestep: out[b,o] = seq[b,T-1,:] . fc_w[o,:] + fc_b[o]
// ---------------------------------------------------------------------------
__global__ void fc_kernel(const __bf16* __restrict__ seq,
                          const float* __restrict__ fw,
                          const float* __restrict__ fb,
                          float* __restrict__ out)
{
    int i = blockIdx.x * blockDim.x + threadIdx.x;  // 0 .. B*DOUT-1
    int b = i >> 7;
    int o = i & (DOUT - 1);
    const __bf16* h = seq + ((size_t)b * TT + (TT - 1)) * HH;
    const float*  w = fw + (size_t)o * HH;
    float s = fb[o];
    for (int k = 0; k < HH; ++k) s += (float)h[k] * w[k];
    out[i] = s;
}

__global__ void reset_counter(unsigned int* c) {
    if (threadIdx.x == 0 && blockIdx.x == 0) *c = 0u;
}

// ---------------------------------------------------------------------------
extern "C" void kernel_launch(void* const* d_in, const int* in_sizes, int n_in,
                              void* d_out, int out_size, void* d_ws, size_t ws_size,
                              hipStream_t stream)
{
    const float* x     = (const float*)d_in[0];   // [B,T,D_IN]
    const float* w_ih0 = (const float*)d_in[1];   // [H,D_IN]
    const float* w_ihr = (const float*)d_in[2];   // [L-1,H,H]
    const float* w_hh  = (const float*)d_in[3];   // [L,H,H]
    const float* b_ih  = (const float*)d_in[4];   // [L,H]
    const float* b_hh  = (const float*)d_in[5];   // [L,H]
    const float* fc_w  = (const float*)d_in[6];   // [D_OUT,H]
    const float* fc_b  = (const float*)d_in[7];   // [D_OUT]
    float* out = (float*)d_out;
    (void)in_sizes; (void)n_in; (void)out_size; (void)ws_size;

    char* ws = (char*)d_ws;
    __bf16* seqA = (__bf16*)(ws);                           // 64 MB
    __bf16* seqB = (__bf16*)(ws + ((size_t)64  << 20));     // 64 MB
    float*  pre  = (float*) (ws + ((size_t)128 << 20));     // 128 MB
    __bf16* wih  = (__bf16*)(ws + ((size_t)256 << 20));     // 2 MB
    __bf16* whh  = (__bf16*)(ws + ((size_t)258 << 20));     // 2 MB
    unsigned int* counter = (unsigned int*)(ws + ((size_t)260 << 20));

    const int M = BB * TT;                                  // 32768 rows
    const size_t lds_bytes = (size_t)RNN_COLS * HH * sizeof(__bf16);  // 256 KB

    auto cvt = [&](const float* in, __bf16* o, int n) {
        int n4 = n >> 2;
        cvt_f32_bf16_x4<<<dim3((n4 + 255) / 256), dim3(256), 0, stream>>>(in, o, n4);
    };

    // ---- Layer 0: K = D_IN = 256 ----
    cvt(x,     seqA, BB * TT * DIN);
    cvt(w_ih0, wih,  HH * DIN);
    cvt(w_hh,  whh,  HH * HH);
    {
        int blocks = (M / 16) * (HH / 64) / 8;              // 4096
        gemm_bf16_bias<<<dim3(blocks), dim3(256), 0, stream>>>(
            seqA, wih, b_ih, b_hh, pre, M, HH, DIN);
    }
    reset_counter<<<dim3(1), dim3(64), 0, stream>>>(counter);
    rnn_scan<<<dim3(RNN_BLOCKS), dim3(RNN_THREADS), lds_bytes, stream>>>(
        pre, whh, seqB, counter);

    // ---- Layers 1..2: K = H = 1024 ----
    for (int l = 1; l < 3; ++l) {
        __bf16* sIn  = (l & 1) ? seqB : seqA;
        __bf16* sOut = (l & 1) ? seqA : seqB;
        cvt(w_ihr + (size_t)(l - 1) * HH * HH, wih, HH * HH);
        cvt(w_hh  + (size_t)l * HH * HH,       whh, HH * HH);
        int blocks = (M / 16) * (HH / 64) / 8;
        gemm_bf16_bias<<<dim3(blocks), dim3(256), 0, stream>>>(
            sIn, wih, b_ih + (size_t)l * HH, b_hh + (size_t)l * HH, pre, M, HH, HH);
        reset_counter<<<dim3(1), dim3(64), 0, stream>>>(counter);
        rnn_scan<<<dim3(RNN_BLOCKS), dim3(RNN_THREADS), lds_bytes, stream>>>(
            pre, whh, sOut, counter);
    }

    // ---- Final FC on last timestep (layer-2 output lives in seqB) ----
    fc_kernel<<<dim3((BB * DOUT) / 256), dim3(256), 0, stream>>>(
        seqB, fc_w, fc_b, out);
}
